// FusedMixtureOfExperts_89223650607340
// MI455X (gfx1250) — matrix operand order repaired and checked by reading, not compiled, not run on hardware
//
#include <hip/hip_runtime.h>
#include <hip/hip_bf16.h>
#include <cstddef>

// ---------------------------------------------------------------------------
// Fused top-2 MoE for MI455X (gfx1250, wave32, WMMA + async-to-LDS DMA).
// Top-2 sparsity: 69 GFLOP bf16 WMMA instead of the reference's dense
// 275 GFLOP fp32. fc kernels use double-buffered GLOBAL_LOAD_ASYNC_TO_LDS_B128
// (ASYNCcnt) so global->LDS DMA overlaps v_wmma_f32_16x16x32_bf16 issue.
// Workspace: ~105 MB (routing + bf16 act + bf16 copies of x/w1/w2).
// ---------------------------------------------------------------------------

#define N_TOK 4096   // B*T
#define DDIM  1024
#define HDIM  2048
#define NEXP  8

typedef __attribute__((ext_vector_type(16))) __bf16 v16bf;
typedef __attribute__((ext_vector_type(8)))  float  v8f;
typedef __attribute__((ext_vector_type(8)))  __bf16 bf16x8;
typedef __attribute__((ext_vector_type(4)))  __bf16 bf16x4;
typedef __attribute__((ext_vector_type(4)))  float  f32x4;

constexpr int LDSROW = 40;                 // 32 bf16 + 8 pad (80B rows, 16B aligned)
constexpr unsigned ABUFB = 64  * LDSROW * 2;   // 5120 B per A buffer
constexpr unsigned BBUFB = 128 * LDSROW * 2;   // 10240 B per B buffer

// Generic->LDS: low 32 bits of the flat address are the LDS byte offset.
__device__ __forceinline__ unsigned lds_off(const void* p) {
  return (unsigned)(size_t)p;
}
// CDNA5 async DMA: copies 16B/lane from global into LDS, tracked by ASYNCcnt.
__device__ __forceinline__ void async_b128(unsigned dst_lds, const __bf16* src) {
  asm volatile("global_load_async_to_lds_b128 %0, %1, off"
               :: "v"(dst_lds), "v"(src) : "memory");
}

// ---------------------------------------------------------------- init ------
__global__ __launch_bounds__(256) void moe_init(float* __restrict__ out,
                                                int* __restrict__ ints) {
  size_t i = (size_t)blockIdx.x * 256 + threadIdx.x;   // grid covers N*D/4
  f32x4 z = {0.f, 0.f, 0.f, 0.f};
  ((f32x4*)out)[i] = z;
  if (blockIdx.x == 0 && threadIdx.x < 3 * NEXP) ints[threadIdx.x] = 0;
}

// -------------------------------------------------------- fp32 -> bf16 ------
__global__ __launch_bounds__(256) void moe_cvt(const float* __restrict__ src,
                                               __bf16* __restrict__ dst) {
  size_t i = ((size_t)blockIdx.x * 256 + threadIdx.x) * 4;
  f32x4 v = *(const f32x4*)(src + i);
  bf16x4 o = { (__bf16)v.x, (__bf16)v.y, (__bf16)v.z, (__bf16)v.w };
  *(bf16x4*)(dst + i) = o;
}

// ---------------------------------------------------------------- gate ------
__global__ __launch_bounds__(256)
void moe_gate(const float* __restrict__ x, const unsigned char* __restrict__ pm,
              const float* __restrict__ gw, float* __restrict__ gate_out,
              int* __restrict__ counts, int* __restrict__ top1,
              int* __restrict__ tok_idx, float* __restrict__ tok_w) {
  const int t = blockIdx.x * 8 + (threadIdx.x >> 5);   // one wave per token
  const int lane = threadIdx.x & 31;
  float acc[NEXP] = {};
  const float* xr = x + (size_t)t * DDIM;
  for (int d = lane; d < DDIM; d += 32) {
    float xv = xr[d];
#pragma unroll
    for (int e = 0; e < NEXP; ++e) acc[e] += xv * gw[e * DDIM + d];
  }
#pragma unroll
  for (int e = 0; e < NEXP; ++e) {
#pragma unroll
    for (int off = 16; off > 0; off >>= 1) acc[e] += __shfl_xor(acc[e], off, 32);
  }
  if (lane == 0) {
    const bool masked = pm[t] != 0;
    float mx = acc[0];
#pragma unroll
    for (int e = 1; e < NEXP; ++e) mx = fmaxf(mx, acc[e]);
    float p[NEXP], s = 0.f;
#pragma unroll
    for (int e = 0; e < NEXP; ++e) { p[e] = expf(acc[e] - mx); s += p[e]; }
    const float inv = 1.f / s;
#pragma unroll
    for (int e = 0; e < NEXP; ++e) {
      p[e] = masked ? 0.f : p[e] * inv;
      gate_out[(size_t)t * NEXP + e] = p[e];
    }
    if (!masked) {
      int i1 = 0; float v1 = p[0];
#pragma unroll
      for (int e = 1; e < NEXP; ++e) if (p[e] > v1) { v1 = p[e]; i1 = e; }
      int i2 = -1; float v2 = -1.f;
#pragma unroll
      for (int e = 0; e < NEXP; ++e)
        if (e != i1 && p[e] > v2) { v2 = p[e]; i2 = e; }
      float den = v1 + v2; if (den <= 0.f) den = 1.f;
      atomicAdd(&top1[i1], 1);
      int pos1 = atomicAdd(&counts[i1], 1);
      tok_idx[i1 * N_TOK + pos1] = t; tok_w[i1 * N_TOK + pos1] = v1 / den;
      int pos2 = atomicAdd(&counts[i2], 1);
      tok_idx[i2 * N_TOK + pos2] = t; tok_w[i2 * N_TOK + pos2] = v2 / den;
    }
  }
}

// -------------------------------------------------------------- prefix ------
__global__ void moe_prefix(const int* __restrict__ counts, int* __restrict__ prefix) {
  if (threadIdx.x == 0) {
    int run = 0;
    for (int e = 0; e < NEXP; ++e) { prefix[e] = run; run += counts[e]; }
  }
}

// ------------------------------------------------ shared fragment helpers ---
// A 16x32 bf16 fragment: lane holds runs [kbase, kbase+8) and [kbase+16, +24)
__device__ __forceinline__ v16bf load_a_frag(const __bf16* sA, int row, int lh) {
  const __bf16* p = sA + row * LDSROW + lh * 8;
  bf16x8 lo = *(const bf16x8*)p;
  bf16x8 hi = *(const bf16x8*)(p + 16);
  return __builtin_shufflevector(lo, hi, 0,1,2,3,4,5,6,7,8,9,10,11,12,13,14,15);
}
// B 32x16 bf16 fragment: lane holds runs [kbase, kbase+8) and [kbase+16, +24) of one column
__device__ __forceinline__ v16bf load_b_frag(const __bf16* sB, int col, int lh) {
  const __bf16* p = sB + col * LDSROW + lh * 8;
  bf16x8 lo = *(const bf16x8*)p;
  bf16x8 hi = *(const bf16x8*)(p + 16);
  return __builtin_shufflevector(lo, hi, 0,1,2,3,4,5,6,7,8,9,10,11,12,13,14,15);
}

// ------------------------------------------------------------------ fc1 -----
// act[m,n] = relu( sum_k Xbf[tok[m],k]*W1bf[e*H+n,k] + b1[e*H+n] )
__global__ __launch_bounds__(256)
void moe_fc1(const __bf16* __restrict__ xbf, const __bf16* __restrict__ w1bf,
             const float* __restrict__ b1, const int* __restrict__ counts,
             const int* __restrict__ prefix, const int* __restrict__ tok_idx,
             __bf16* __restrict__ act) {
  const int e = blockIdx.z;
  const int cnt = counts[e];
  const int m0 = blockIdx.x * 64;
  if (m0 >= cnt) return;
  const int n0 = blockIdx.y * 128;
  const int pref = prefix[e];

  __shared__ __align__(16) __bf16 sA[2][64 * LDSROW];
  __shared__ __align__(16) __bf16 sB[2][128 * LDSROW];

  const int tid  = threadIdx.x;
  const int wave = tid >> 5, lane = tid & 31;
  const int lr = lane & 15, lh = lane >> 4;
  const int wm = (wave >> 2) * 32, wn = (wave & 3) * 32;

  // A staging: 256 chunks of 8 bf16 (64 rows x 4) -> one async b128 per lane
  const int ara = tid >> 2, aca = (tid & 3) * 8;
  int rowa = m0 + ara; if (rowa >= cnt) rowa = cnt - 1;
  const __bf16* ag = xbf + (size_t)tok_idx[e * N_TOK + rowa] * DDIM + aca;
  const unsigned al = lds_off(&sA[0][ara * LDSROW + aca]);
  // B staging: 512 chunks (128 rows x 4) -> two async b128 per lane
  const __bf16* bg[2]; unsigned bl[2];
#pragma unroll
  for (int i = 0; i < 2; ++i) {
    int idx = tid * 2 + i, r = idx >> 2, c8 = (idx & 3) * 8;
    bg[i] = w1bf + (size_t)(e * HDIM + n0 + r) * DDIM + c8;
    bl[i] = lds_off(&sB[0][r * LDSROW + c8]);
  }
  auto issue = [&](int kk, unsigned buf) {
    async_b128(al + buf * ABUFB, ag + kk * 32);
    async_b128(bl[0] + buf * BBUFB, bg[0] + kk * 32);
    async_b128(bl[1] + buf * BBUFB, bg[1] + kk * 32);
  };

  v8f acc[2][2] = {};
  issue(0, 0);
  for (int kk = 0; kk < DDIM / 32; ++kk) {
    const int buf = kk & 1;
    if (kk + 1 < DDIM / 32) {
      issue(kk + 1, buf ^ 1);
      asm volatile("s_wait_asynccnt 0x3" ::: "memory");  // group kk landed
    } else {
      asm volatile("s_wait_asynccnt 0x0" ::: "memory");
    }
    __syncthreads();
    v16bf aF[2], bF[2];
#pragma unroll
    for (int mi = 0; mi < 2; ++mi) aF[mi] = load_a_frag(sA[buf], wm + mi * 16 + lr, lh);
#pragma unroll
    for (int ni = 0; ni < 2; ++ni) bF[ni] = load_b_frag(sB[buf], wn + ni * 16 + lr, lh);
#pragma unroll
    for (int mi = 0; mi < 2; ++mi)
#pragma unroll
      for (int ni = 0; ni < 2; ++ni)
        acc[mi][ni] = __builtin_amdgcn_wmma_f32_16x16x32_bf16(
            false, aF[mi], false, bF[ni], (short)0, acc[mi][ni], false, false);
    __syncthreads();
  }

  const bool full = (m0 + 64 <= cnt);
  float b1v[2];
#pragma unroll
  for (int ni = 0; ni < 2; ++ni) b1v[ni] = b1[e * HDIM + n0 + wn + ni * 16 + lr];
#pragma unroll
  for (int mi = 0; mi < 2; ++mi)
#pragma unroll
    for (int r = 0; r < 8; ++r) {
      int lrow = wm + mi * 16 + lh * 8 + r;
      if (full || (m0 + lrow < cnt)) {
        size_t base = (size_t)(pref + m0 + lrow) * HDIM + n0 + wn + lr;
#pragma unroll
        for (int ni = 0; ni < 2; ++ni) {
          float c = acc[mi][ni][r] + b1v[ni];
          act[base + ni * 16] = (__bf16)(c > 0.f ? c : 0.f);
        }
      }
    }
}

// ------------------------------------------------------------------ fc2 -----
// out[tok[m],n] += g[m] * ( sum_k act[m,k]*W2bf[e*D+n,k] + b2[e*D+n] )
__global__ __launch_bounds__(256)
void moe_fc2(const __bf16* __restrict__ act, const __bf16* __restrict__ w2bf,
             const float* __restrict__ b2, const int* __restrict__ counts,
             const int* __restrict__ prefix, const int* __restrict__ tok_idx,
             const float* __restrict__ tok_w, float* __restrict__ out) {
  const int e = blockIdx.z;
  const int cnt = counts[e];
  const int m0 = blockIdx.x * 64;
  if (m0 >= cnt) return;
  const int n0 = blockIdx.y * 128;
  const int pref = prefix[e];

  __shared__ __align__(16) __bf16 sA[2][64 * LDSROW];
  __shared__ __align__(16) __bf16 sB[2][128 * LDSROW];

  const int tid  = threadIdx.x;
  const int wave = tid >> 5, lane = tid & 31;
  const int lr = lane & 15, lh = lane >> 4;
  const int wm = (wave >> 2) * 32, wn = (wave & 3) * 32;

  const int ara = tid >> 2, aca = (tid & 3) * 8;
  int rowa = m0 + ara; if (rowa >= cnt) rowa = cnt - 1;
  const __bf16* ag = act + (size_t)(pref + rowa) * HDIM + aca;
  const unsigned al = lds_off(&sA[0][ara * LDSROW + aca]);
  const __bf16* bg[2]; unsigned bl[2];
#pragma unroll
  for (int i = 0; i < 2; ++i) {
    int idx = tid * 2 + i, r = idx >> 2, c8 = (idx & 3) * 8;
    bg[i] = w2bf + (size_t)(e * DDIM + n0 + r) * HDIM + c8;
    bl[i] = lds_off(&sB[0][r * LDSROW + c8]);
  }
  auto issue = [&](int kk, unsigned buf) {
    async_b128(al + buf * ABUFB, ag + kk * 32);
    async_b128(bl[0] + buf * BBUFB, bg[0] + kk * 32);
    async_b128(bl[1] + buf * BBUFB, bg[1] + kk * 32);
  };

  v8f acc[2][2] = {};
  issue(0, 0);
  for (int kk = 0; kk < HDIM / 32; ++kk) {
    const int buf = kk & 1;
    if (kk + 1 < HDIM / 32) {
      issue(kk + 1, buf ^ 1);
      asm volatile("s_wait_asynccnt 0x3" ::: "memory");
    } else {
      asm volatile("s_wait_asynccnt 0x0" ::: "memory");
    }
    __syncthreads();
    v16bf aF[2], bF[2];
#pragma unroll
    for (int mi = 0; mi < 2; ++mi) aF[mi] = load_a_frag(sA[buf], wm + mi * 16 + lr, lh);
#pragma unroll
    for (int ni = 0; ni < 2; ++ni) bF[ni] = load_b_frag(sB[buf], wn + ni * 16 + lr, lh);
#pragma unroll
    for (int mi = 0; mi < 2; ++mi)
#pragma unroll
      for (int ni = 0; ni < 2; ++ni)
        acc[mi][ni] = __builtin_amdgcn_wmma_f32_16x16x32_bf16(
            false, aF[mi], false, bF[ni], (short)0, acc[mi][ni], false, false);
    __syncthreads();
  }

  const bool full = (m0 + 64 <= cnt);
  float b2v[2];
#pragma unroll
  for (int ni = 0; ni < 2; ++ni) b2v[ni] = b2[e * DDIM + n0 + wn + ni * 16 + lr];
#pragma unroll
  for (int mi = 0; mi < 2; ++mi)
#pragma unroll
    for (int r = 0; r < 8; ++r) {
      int lrow = wm + mi * 16 + lh * 8 + r;
      if (full || (m0 + lrow < cnt)) {
        int li = e * N_TOK + m0 + lrow;
        int t = tok_idx[li];
        float g = tok_w[li];
        float* orow = out + (size_t)t * DDIM + n0 + wn + lr;
#pragma unroll
        for (int ni = 0; ni < 2; ++ni) {
          float c = acc[mi][ni][r] + b2v[ni];
          // exactly 2 contributions per element -> commutative -> deterministic
          __hip_atomic_fetch_add(orow + ni * 16, g * c,
                                 __ATOMIC_RELAXED, __HIP_MEMORY_SCOPE_AGENT);
        }
      }
    }
}

// -------------------------------------------------------------- lb loss -----
__global__ __launch_bounds__(256)
void moe_lb(const float* __restrict__ gate_out, const unsigned char* __restrict__ pm,
            const int* __restrict__ top1, float* __restrict__ lb_out) {
  __shared__ float red[256];
  __shared__ int   redi[256];
  __shared__ float colsum[NEXP];
  const int tid = threadIdx.x;
  int c = 0;
  for (int t = tid; t < N_TOK; t += 256) c += (pm[t] == 0);
  redi[tid] = c; __syncthreads();
  for (int s = 128; s > 0; s >>= 1) {
    if (tid < s) redi[tid] += redi[tid + s];
    __syncthreads();
  }
  for (int e = 0; e < NEXP; ++e) {
    float s = 0.f;
    for (int t = tid; t < N_TOK; t += 256) s += gate_out[(size_t)t * NEXP + e];
    red[tid] = s; __syncthreads();
    for (int s2 = 128; s2 > 0; s2 >>= 1) {
      if (tid < s2) red[tid] += red[tid + s2];
      __syncthreads();
    }
    if (tid == 0) colsum[e] = red[0];
    __syncthreads();
  }
  if (tid == 0) {
    float nt = (float)redi[0];
    float lb = 0.f;
    for (int e = 0; e < NEXP; ++e)
      lb += ((float)top1[e] / nt) * (colsum[e] / nt);
    lb_out[0] = (float)NEXP * lb;
  }
}

// ---------------------------------------------------------------------------
extern "C" void kernel_launch(void* const* d_in, const int* in_sizes, int n_in,
                              void* d_out, int out_size, void* d_ws, size_t ws_size,
                              hipStream_t stream) {
  const float* x      = (const float*)d_in[0];
  const unsigned char* pm = (const unsigned char*)d_in[1];  // jnp bool = 1 byte
  const float* gate_w = (const float*)d_in[2];
  const float* fc1_w  = (const float*)d_in[3];
  const float* fc1_b  = (const float*)d_in[4];
  const float* fc2_w  = (const float*)d_in[5];
  const float* fc2_b  = (const float*)d_in[6];
  float* out = (float*)d_out;                       // [N*D | N*E | 1]
  float* gate_out = out + (size_t)N_TOK * DDIM;
  float* lb_out   = gate_out + (size_t)N_TOK * NEXP;

  // workspace layout (~105 MB)
  char* ws = (char*)d_ws;
  int*   counts  = (int*)ws;                         // E + E + E ints
  int*   prefix  = counts + NEXP;
  int*   top1    = prefix + NEXP;
  int*   tok_idx = (int*)(ws + 256);                             // E*N ints
  float* tok_w   = (float*)(ws + 256 + NEXP * N_TOK * 4);        // E*N floats
  constexpr size_t ACT_OFF = 256 + (size_t)NEXP * N_TOK * 8;     // 262400
  constexpr size_t ACT_SZ  = (size_t)2 * N_TOK * HDIM * 2;       // 32 MB
  constexpr size_t XBF_OFF = ACT_OFF + ACT_SZ;
  constexpr size_t XBF_SZ  = (size_t)N_TOK * DDIM * 2;           // 8 MB
  constexpr size_t W1_OFF  = XBF_OFF + XBF_SZ;
  constexpr size_t W1_SZ   = (size_t)NEXP * HDIM * DDIM * 2;     // 32 MB
  constexpr size_t W2_OFF  = W1_OFF + W1_SZ;
  __bf16* act  = (__bf16*)(ws + ACT_OFF);
  __bf16* xbf  = (__bf16*)(ws + XBF_OFF);
  __bf16* w1bf = (__bf16*)(ws + W1_OFF);
  __bf16* w2bf = (__bf16*)(ws + W2_OFF);

  moe_init<<<(N_TOK * DDIM) / (256 * 4), 256, 0, stream>>>(out, counts);
  moe_cvt<<<(N_TOK * DDIM) / (256 * 4), 256, 0, stream>>>(x, xbf);
  moe_cvt<<<(NEXP * HDIM * DDIM) / (256 * 4), 256, 0, stream>>>(fc1_w, w1bf);
  moe_cvt<<<(NEXP * DDIM * HDIM) / (256 * 4), 256, 0, stream>>>(fc2_w, w2bf);
  moe_gate<<<N_TOK / 8, 256, 0, stream>>>(x, pm, gate_w, gate_out,
                                          counts, top1, tok_idx, tok_w);
  moe_prefix<<<1, 32, 0, stream>>>(counts, prefix);
  moe_fc1<<<dim3(N_TOK / 64, HDIM / 128, NEXP), 256, 0, stream>>>(
      xbf, w1bf, fc1_b, counts, prefix, tok_idx, act);
  moe_fc2<<<dim3(N_TOK / 64, DDIM / 128, NEXP), 256, 0, stream>>>(
      act, w2bf, fc2_b, counts, prefix, tok_idx, tok_w, out);
  moe_lb<<<1, 256, 0, stream>>>(gate_out, pm, top1, lb_out);
}
